// EKFPredRNN_68212670595231
// MI455X (gfx1250) — matrix operand order repaired and checked by reading, not compiled, not run on hardware
//
#include <hip/hip_runtime.h>
#include <cstdint>
#include <cstddef>

typedef __attribute__((ext_vector_type(16))) _Float16 v16h;
typedef __attribute__((ext_vector_type(8)))  float    v8f;

#define ACT_NONE  0
#define ACT_LRELU 1
#define ACT_TANH  2
#define ACT_SIG   3

#if __has_builtin(__builtin_amdgcn_tensor_load_to_lds)
#define USE_TDM 1
#else
#define USE_TDM 0
#endif

#if USE_TDM
typedef __attribute__((ext_vector_type(4))) unsigned v4u;
typedef __attribute__((ext_vector_type(8))) int      v8i;
typedef __attribute__((ext_vector_type(4))) int      v4i;
// 2-D TDM tile load: tile1 rows x tile0 f16 elements, row stride stride0
// (data_size units), into packed LDS at `lds`. OOB reads (past td0/td1)
// return zero, which subsumes K-tail / Cout-tail zero fill.
__device__ __forceinline__ void tdm_load_2d(void* lds, const _Float16* g,
                                            unsigned td0, unsigned td1,
                                            unsigned tile0, unsigned tile1,
                                            unsigned stride0) {
  unsigned ldsb = (unsigned)(uintptr_t)lds;                 // low 32 = LDS offset
  unsigned long long ga = (unsigned long long)(uintptr_t)g; // 57-bit global addr
  v4u g0;
  g0[0] = 1u;                                   // count=1, user D#
  g0[1] = ldsb;                                 // lds_addr
  g0[2] = (unsigned)ga;                         // global_addr[31:0]
  g0[3] = (unsigned)((ga >> 32) & 0x01FFFFFFull) | (2u << 30);  // addr[56:32] | type=2
  v8i g1;
  g1[0] = (int)0x00010000u;                     // data_size=1 (2 bytes), no pad/iter
  g1[1] = (int)((td0 & 0xFFFFu) << 16);         // tensor_dim0[15:0] @ bits 63:48
  g1[2] = (int)((td0 >> 16) | ((td1 & 0xFFFFu) << 16));   // td0 hi | td1 lo
  g1[3] = (int)((td1 >> 16) | (tile0 << 16));   // td1 hi | tile_dim0
  g1[4] = (int)tile1;                           // tile_dim1 (tile_dim2=0)
  g1[5] = (int)stride0;                         // tensor_dim0_stride[31:0]
  g1[6] = 0; g1[7] = 0;                         // stride hi / dim1_stride = 0
  v4i gz; gz[0] = 0; gz[1] = 0; gz[2] = 0; gz[3] = 0;
#if __clang_major__ >= 23
  v8i gz8; for (int i = 0; i < 8; ++i) gz8[i] = 0;
  __builtin_amdgcn_tensor_load_to_lds(g0, g1, gz, gz, gz8, 0);
#else
  __builtin_amdgcn_tensor_load_to_lds(g0, g1, gz, gz, 0);
#endif
}
#endif

__device__ __forceinline__ float sigf(float x) { return 1.0f / (1.0f + __expf(-x)); }
__device__ __forceinline__ float actf(float x, int act, float s) {
  if (act == ACT_LRELU) return x >= 0.0f ? x : s * x;
  if (act == ACT_TANH)  return tanhf(x);
  if (act == ACT_SIG)   return sigf(x);
  return x;
}

__global__ void cvt16(const float* s, _Float16* d, int n) {
  int i = blockIdx.x * blockDim.x + threadIdx.x;
  if (i < n) d[i] = (_Float16)s[i];
}

// ---------------------------------------------------------------------------
// Implicit-GEMM conv (NCHW, OIHW f16 weights). Block = 128 thr = 4 wave32.
// Block tile M=64 pixels x N=64 out-ch; K staged 64 per barrier (2 WMMA
// k-steps). A-tile: per-thread im2col f32->f16 with pixel decomposition
// hoisted out of the K loop (KS template constant, pow2 spatial shifts).
// B-tile: Tensor Data Mover 2-D descriptor load (TENSORcnt) while VALU does
// the A conversion. LDS holds plain k order; each lane reads its v16h fragment as
// contiguous b128 DS loads (k = 16*(lane/16) + s, same order for A and B).
// ---------------------------------------------------------------------------
struct ConvArgs {
  const float* inA; const float* inB; const float* inC;
  int Ca, Cb, Cc, Cin;
  const _Float16* w; const float* bias; float* out;
  int B, Hin, Win, Cout, hwshift, wshift, stride, pad;
  int act; float slope;
};

template <int KS>
__global__ __launch_bounds__(128) void conv_wmma(ConvArgs a) {
  __shared__ _Float16 As[64][64];   // [m_local][k]
  __shared__ _Float16 Bs[64][64];   // [n_local][k] (packed tile: TDM target)
  constexpr int KS2 = KS * KS;
  const int t = threadIdx.x;
  const int lane = t & 31, wave = t >> 5;
  const int Ktot = a.Cin * KS2;
  const int HWo  = 1 << a.hwshift;
  const int M    = a.B << a.hwshift;
  const int tileM = blockIdx.x * 64;
  const int tileN = blockIdx.y * 64;
  // --- per-thread staging coordinates (fixed across K loop) ---
  const int arow = t >> 1;               // LDS row this thread fills
  const int akk0 = (t & 1) << 5;         // k sub-offset {0,32}
  const int mg   = tileM + arow;
  const int mb   = mg >> a.hwshift;      // batch
  const int mp   = mg & (HWo - 1);
  const int oy   = mp >> a.wshift;
  const int ox   = mp & ((1 << a.wshift) - 1);
  const int iy0  = oy * a.stride - a.pad;
  const int ix0  = ox * a.stride - a.pad;
  const bool mok = mg < M;
#if !USE_TDM
  const int co   = tileN + arow;
  const bool cok = co < a.Cout;
  const _Float16* wrow = a.w + (size_t)co * Ktot + akk0;
#endif

  v8f acc[4];
  v8f zero = {};
  for (int i = 0; i < 4; ++i) acc[i] = zero;

  const int nK = (Ktot + 63) >> 6;
  for (int kc = 0; kc < nK; ++kc) {
    const int kb = (kc << 6) + akk0;
    // ---- stage B: TDM moves the 64x64-f16 weight tile while VALU does A ----
#if USE_TDM
    if (t == 0) {   // TDM ignores EXEC; one issue per block
      tdm_load_2d(&Bs[0][0], a.w + (size_t)tileN * Ktot + (kc << 6),
                  (unsigned)(Ktot - (kc << 6)), (unsigned)(a.Cout - tileN),
                  64u, 64u, (unsigned)Ktot);
    }
#endif
    // ---- stage A: this thread's 32 im2col elements (f32 -> f16) ----
#pragma unroll
    for (int i = 0; i < 32; ++i) {
      int k = kb + i;
      float v = 0.0f;
      if (mok && k < Ktot) {
        int cin = k / KS2;               // constant divisor
        int tap = k - cin * KS2;
        int iy = iy0 + tap / KS;
        int ix = ix0 + tap % KS;
        if (iy >= 0 && iy < a.Hin && ix >= 0 && ix < a.Win) {
          const float* src = a.inA; int cl = cin; int Cs = a.Ca;
          if (cin >= a.Ca + a.Cb)      { src = a.inC; cl = cin - a.Ca - a.Cb; Cs = a.Cc; }
          else if (cin >= a.Ca)        { src = a.inB; cl = cin - a.Ca;        Cs = a.Cb; }
          v = src[((mb * Cs + cl) * a.Hin + iy) * a.Win + ix];
        }
      }
      int kk = akk0 + i;
      int pos = kk;
      As[arow][pos] = (_Float16)v;
    }
#if USE_TDM
    if (t == 0) __builtin_amdgcn_s_wait_tensorcnt(0);  // B tile landed in LDS
#else
    if (cok && kb + 32 <= Ktot) {
      *(v16h*)&Bs[arow][akk0]      = *(const v16h*)(wrow + (kc << 6));
      *(v16h*)&Bs[arow][akk0 + 16] = *(const v16h*)(wrow + (kc << 6) + 16);
    } else {
      for (int i = 0; i < 32; ++i) {
        int k = kb + i;
        Bs[arow][akk0 + i] = (cok && k < Ktot) ? wrow[(kc << 6) + i] : (_Float16)0.0f;
      }
    }
    if (t == 0 && kc + 1 < nK)
      __builtin_prefetch(wrow + ((kc + 1) << 6), 0, 0);  // global_prefetch_b8
#endif
    __syncthreads();
    const int hi = lane >> 4;
    const int mrow = (wave << 4) + (lane & 15);
#pragma unroll
    for (int kh = 0; kh < 2; ++kh) {
      v16h afrag = *(const v16h*)&As[mrow][(kh << 5) + (hi << 4)];
#pragma unroll
      for (int nf = 0; nf < 4; ++nf) {
        v16h bfrag = *(const v16h*)&Bs[(nf << 4) + (lane & 15)][(kh << 5) + (hi << 4)];
        acc[nf] = __builtin_amdgcn_wmma_f32_16x16x32_f16(
            false, afrag, false, bfrag, (short)0, acc[nf], false, false);
      }
    }
    __syncthreads();   // next chunk's TDM/store may not touch Bs before this
  }
  // ---- epilogue: C layout VGPR r -> M = r + hi*8, N = lane&15 ----
  const int hi = lane >> 4, nl = lane & 15;
  for (int nf = 0; nf < 4; ++nf) {
    int n = tileN + (nf << 4) + nl;
    if (n >= a.Cout) continue;
#pragma unroll
    for (int r = 0; r < 8; ++r) {
      int om = tileM + (wave << 4) + (hi << 3) + r;
      if (om >= M) continue;
      int b = om >> a.hwshift, p = om & (HWo - 1);
      float v = acc[nf][r] + a.bias[n];
      a.out[((b * a.Cout + n) << a.hwshift) + p] = actf(v, a.act, a.slope);
    }
  }
}

// ---------------------------------------------------------------------------
// Row GEMM for M=16 batch rows: Y(16,N) = X(16,K) @ W(N,K)^T + bias (W in f16).
// ---------------------------------------------------------------------------
__global__ __launch_bounds__(128)
void rowgemm_wmma(const float* X, const _Float16* W, const float* bias, float* Y,
                  int K, int N, int act, float slope) {
  __shared__ _Float16 As[16][32];
  __shared__ _Float16 Bs[64][32];
  const int t = threadIdx.x;
  const int lane = t & 31, wave = t >> 5;
  const int tileN = blockIdx.x * 64;
  v8f acc = {};
  const int nK = (K + 31) >> 5;
  for (int kc = 0; kc < nK; ++kc) {
    const int kbase = kc << 5;
#if USE_TDM
    if (t == 0) {
      tdm_load_2d(&Bs[0][0], W + (size_t)tileN * K + kbase,
                  (unsigned)(K - kbase), (unsigned)(N - tileN),
                  32u, 64u, (unsigned)K);
    }
#endif
    for (int i = 0; i < 4; ++i) {
      int idx = t + (i << 7);           // 0..511
      int row = idx >> 5, kk = idx & 31;
      int k = kbase + kk;
      float v = (k < K) ? X[row * K + k] : 0.0f;
      int pos = kk;
      As[row][pos] = (_Float16)v;
    }
#if USE_TDM
    if (t == 0) __builtin_amdgcn_s_wait_tensorcnt(0);
#else
    {   // B stage: contiguous f16 copy
      int n = t >> 1, kk0 = (t & 1) << 4;
      int co = tileN + n, kb = kbase + kk0;
      if (co < N && kb + 16 <= K) {
        *(v16h*)&Bs[n][kk0] = *(const v16h*)(W + (size_t)co * K + kb);
      } else {
        for (int i = 0; i < 16; ++i) {
          int k = kb + i;
          Bs[n][kk0 + i] = (co < N && k < K) ? W[(size_t)co * K + k] : (_Float16)0.0f;
        }
      }
    }
#endif
    __syncthreads();
    const int hi = lane >> 4;
    v16h afrag = *(const v16h*)&As[lane & 15][hi << 4];
    v16h bfrag = *(const v16h*)&Bs[(wave << 4) + (lane & 15)][hi << 4];
    acc = __builtin_amdgcn_wmma_f32_16x16x32_f16(
        false, afrag, false, bfrag, (short)0, acc, false, false);
    __syncthreads();
  }
  const int hi = lane >> 4, nl = lane & 15;
  int n = tileN + (wave << 4) + nl;
  if (n < N) {
#pragma unroll
    for (int r = 0; r < 8; ++r) {
      int mrow = (hi << 3) + r;         // batch row
      float v = acc[r] + bias[n];
      Y[mrow * N + n] = actf(v, act, slope);
    }
  }
}

// ------------------------- small / elementwise kernels ----------------------
__global__ void fillz(float* p, int n) {
  int i = blockIdx.x * blockDim.x + threadIdx.x;
  if (i < n) p[i] = 0.0f;
}
__global__ void initP(float* p, int n) {  // 0.1 * I, 32x32 per batch
  int i = blockIdx.x * blockDim.x + threadIdx.x;
  if (i >= n) return;
  int r = (i >> 5) & 31, cidx = i & 31;
  p[i] = (r == cidx) ? 0.1f : 0.0f;
}
__global__ void gatherf(const float* x, float* o, int t) {  // x[:,t] -> (16,1,64,64)
  int i = blockIdx.x * blockDim.x + threadIdx.x;
  if (i >= 16 * 4096) return;
  int b = i >> 12, p = i & 4095;
  o[i] = x[(b * 7 + t) * 4096 + p];
}
__global__ void scat(const float* pred, float* out, int t) {
  int i = blockIdx.x * blockDim.x + threadIdx.x;
  if (i >= 16 * 4096) return;
  int b = i >> 12, p = i & 4095;
  out[(b * 7 + t) * 4096 + p] = pred[i];
}
__global__ void addb(float* x, const float* e, int n) {  // x += bcast(e), per=65536
  int i = blockIdx.x * blockDim.x + threadIdx.x;
  if (i < n) x[i] += e[i & 65535];
}
__global__ void axpy(float* x, const float* y, float a, int n) {
  int i = blockIdx.x * blockDim.x + threadIdx.x;
  if (i < n) x[i] += a * y[i];
}
__global__ void ew_sub(const float* a, const float* b, float* o, int n) {
  int i = blockIdx.x * blockDim.x + threadIdx.x;
  if (i < n) o[i] = a[i] - b[i];
}
__global__ void lstm_cm(const float* g4, float* c, int n) {  // n = 16*64*1024
  int i = blockIdx.x * blockDim.x + threadIdx.x;
  if (i >= n) return;
  int b = i >> 16, r = i & 65535;
  const float* gb = g4 + (b << 18);
  float ig = gb[r], fg = gb[65536 + r], gg = gb[131072 + r];
  c[i] = sigf(fg) * c[i] + sigf(ig) * tanhf(gg);
}
__global__ void lstm_hm(const float* g4, const float* wmo, const float* c,
                        float* m, float* hl, int n) {
  int i = blockIdx.x * blockDim.x + threadIdx.x;
  if (i >= n) return;
  int b = i >> 16, r = i & 65535;
  float og = g4[(b << 18) + 196608 + r];
  float tc = tanhf(c[i]);
  m[i]  = sigf(wmo[i]) * tc;
  hl[i] = sigf(og) * tc;
}
__global__ void pool2(const float* X, float* Y, int n) {  // 32x32 -> 16x16 avg
  int i = blockIdx.x * blockDim.x + threadIdx.x;
  if (i >= n) return;
  int ox = i & 15, tmp = i >> 4;
  int oy = tmp & 15; int bc = tmp >> 4;
  const float* xp = X + (bc * 32 + oy * 2) * 32 + ox * 2;
  Y[i] = 0.25f * (xp[0] + xp[1] + xp[32] + xp[33]);
}
__global__ void bilin(const float* X, float* Y, int n) {  // 16x16 -> 32x32
  int i = blockIdx.x * blockDim.x + threadIdx.x;
  if (i >= n) return;
  int ox = i & 31, oy = (i >> 5) & 31, bc = i >> 10;
  float sy = oy * 0.5f - 0.25f, sx = ox * 0.5f - 0.25f;
  int y0 = (int)floorf(sy), x0 = (int)floorf(sx);
  float fy = sy - (float)y0, fx = sx - (float)x0;
  int y0c = min(max(y0, 0), 15), y1c = min(max(y0 + 1, 0), 15);
  int x0c = min(max(x0, 0), 15), x1c = min(max(x0 + 1, 0), 15);
  const float* xp = X + bc * 256;
  float v00 = xp[y0c * 16 + x0c], v01 = xp[y0c * 16 + x1c];
  float v10 = xp[y1c * 16 + x0c], v11 = xp[y1c * 16 + x1c];
  Y[i] = (1 - fy) * ((1 - fx) * v00 + fx * v01) + fy * ((1 - fx) * v10 + fx * v11);
}
__global__ void combineh(const float* hl, const float* d, float* h,
                         const float* kfa, const float* rw, int n) {
  int i = blockIdx.x * blockDim.x + threadIdx.x;
  if (i >= n) return;
  float a = sigf(kfa[0]), r = sigf(rw[0]);
  h[i] = hl[i] * (1.0f - a) + d[i] * a + r * h[i];
}
__global__ void layernorm_rows(const float* X, const float* g, const float* be,
                               float* Y, int Nw, int postLrelu, float slope) {
  int row = blockIdx.x, t = threadIdx.x;
  __shared__ float red[128];
  float v = (t < Nw) ? X[row * Nw + t] : 0.0f;
  red[t] = v; __syncthreads();
  for (int s = 64; s > 0; s >>= 1) { if (t < s) red[t] += red[t + s]; __syncthreads(); }
  float mu = red[0] / (float)Nw; __syncthreads();
  float d = (t < Nw) ? v - mu : 0.0f;
  red[t] = d * d; __syncthreads();
  for (int s = 64; s > 0; s >>= 1) { if (t < s) red[t] += red[t + s]; __syncthreads(); }
  float inv = rsqrtf(red[0] / (float)Nw + 1e-5f);
  if (t < Nw) {
    float o = d * inv * g[t] + be[t];
    if (postLrelu) o = o >= 0.0f ? o : slope * o;
    Y[row * Nw + t] = o;
  }
}
// --------------------------- Kalman-filter algebra --------------------------
__global__ __launch_bounds__(1024) void fm_build(const float* Fl, const float* Fh, float* Fm) {
  int t = threadIdx.x, i = t >> 5, j = t & 31;
  float s = (i == j) ? 1.0f : 0.0f;
  for (int k = 0; k < 24; ++k) s += Fl[i * 24 + k] * Fh[k * 32 + j];
  Fm[i * 32 + j] = s;
}
__global__ __launch_bounds__(1024)
void bmm32(const float* A, int sA, const float* B, int sB, float* C, int transB, int addC) {
  int b = blockIdx.x, t = threadIdx.x, i = t >> 5, j = t & 31;
  const float* Ab = A + b * sA;
  const float* Bb = B + b * sB;
  float s = 0.0f;
  for (int k = 0; k < 32; ++k)
    s += Ab[i * 32 + k] * (transB ? Bb[j * 32 + k] : Bb[k * 32 + j]);
  int o = b * 1024 + i * 32 + j;
  if (addC) C[o] += s; else C[o] = s;
}
__global__ void add_diag(float* Mtx, const float* logv, float eps) {
  int b = blockIdx.x, i = threadIdx.x;
  Mtx[b * 1024 + i * 33] += __expf(logv[i]) + eps;
}
__global__ __launch_bounds__(1024) void inv32(const float* S, float* Inv) {
  __shared__ float Aug[32][64];
  __shared__ float fcol[32];
  __shared__ float pinv;
  int b = blockIdx.x, t = threadIdx.x, i = t >> 5, j = t & 31;
  Aug[i][j] = S[b * 1024 + i * 32 + j];
  Aug[i][32 + j] = (i == j) ? 1.0f : 0.0f;
  __syncthreads();
  for (int p = 0; p < 32; ++p) {
    if (t == 0) pinv = 1.0f / Aug[p][p];
    __syncthreads();
    if (i == p) { Aug[p][j] *= pinv; Aug[p][32 + j] *= pinv; }
    __syncthreads();
    if (j == 0) fcol[i] = (i == p) ? 0.0f : Aug[i][p];
    __syncthreads();
    float f = fcol[i];
    if (i != p) {
      Aug[i][j]      -= f * Aug[p][j];
      Aug[i][32 + j] -= f * Aug[p][32 + j];
    }
    __syncthreads();
  }
  Inv[b * 1024 + i * 32 + j] = Aug[i][32 + j];
}
__global__ void kf_pred(const float* Fm, const float* st, const float* nl2,
                        const float* nlw, float* sp) {
  int b = blockIdx.x, i = threadIdx.x;
  float s = 0.0f;
  for (int k = 0; k < 32; ++k) s += Fm[i * 32 + k] * st[b * 32 + k];
  sp[b * 32 + i] = s + sigf(nlw[0]) * nl2[b * 32 + i];
}
__global__ void kf_supd(const float* sp, const float* Km, const float* innov, float* so) {
  int b = blockIdx.x, i = threadIdx.x;
  float s = sp[b * 32 + i];
  for (int j = 0; j < 32; ++j) s += Km[b * 1024 + i * 32 + j] * innov[b * 32 + j];
  so[b * 32 + i] = s;
}
__global__ void ew_t1(const float* K, float* T, int n) {  // T = I - K
  int i = blockIdx.x * blockDim.x + threadIdx.x;
  if (i >= n) return;
  int r = (i & 1023) >> 5, cidx = i & 31;
  T[i] = ((r == cidx) ? 1.0f : 0.0f) - K[i];
}
__global__ void ew_kr(const float* K, const float* logR, float* KR, int n) {
  int i = blockIdx.x * blockDim.x + threadIdx.x;
  if (i >= n) return;
  KR[i] = K[i] * __expf(logR[i & 31]);
}
// ConvTranspose2d: (16,64,32,32) -> (16,32,64,64), kernel 4, stride 2, pad 1 eq.
__global__ void convT_d2(const float* X, const float* W, const float* bias,
                         float* Y, float slope) {
  int i = blockIdx.x * blockDim.x + threadIdx.x;
  if (i >= 16 * 32 * 64 * 64) return;
  int ox = i & 63; int tmp = i >> 6;
  int oy = tmp & 63; tmp >>= 6;
  int co = tmp & 31; int b = tmp >> 5;
  float s = bias[co];
  for (int ky = 0; ky < 4; ++ky) {
    int py = oy + ky - 2;
    if (py < 0 || py > 62 || (py & 1)) continue;
    int iy = py >> 1;
    for (int kx = 0; kx < 4; ++kx) {
      int px = ox + kx - 2;
      if (px < 0 || px > 62 || (px & 1)) continue;
      int ix = px >> 1;
      const float* xr = X + (b * 64 * 32 + iy) * 32 + ix;
      const float* wr = W + co * 16 + (3 - ky) * 4 + (3 - kx);
      for (int ci = 0; ci < 64; ++ci)
        s += xr[ci * 1024] * wr[ci * 512];
    }
  }
  Y[i] = s >= 0.0f ? s : slope * s;
}

// ============================== host orchestration ==========================
#define GRID(n) dim3((unsigned)(((n) + 255) / 256))

extern "C" void kernel_launch(void* const* d_in, const int* in_sizes, int n_in,
                              void* d_out, int out_size, void* d_ws, size_t ws_size,
                              hipStream_t stream) {
  (void)in_sizes; (void)n_in; (void)out_size; (void)ws_size;
  auto P = [&](int i) -> const float* { return (const float*)d_in[i]; };
  // param index map (setup_inputs insertion order, depth-first):
  // 0:x 1:e1w 2:e1b 3:e2w 4:e2b 5:time_emb 6:flow_w0 7:flow_w1 8:flow_b0 9:flow_b1
  // 10+27*l: wg bg wm bm enc_w enc_b enc_lw enc_lb enc_g enc_beta dec_lw dec_lb
  //          dec_w dec_b kf_alpha res_w F_low F_high nl_w1 nl_b1 nl_g nl_beta
  //          nl_w2 nl_b2 nl_weight log_Q log_R
  // 91:d1w 92:d1b 93:d2w 94:d2b 95:d3w 96:d3b
  const int CELL0 = 10, CN = 27;
  const float* x    = P(0);
  const float* temb = P(5);
  float* out = (float*)d_out;

  const int NF = 16 * 64 * 1024;     // one (B,64,32,32) tensor
  char* base = (char*)d_ws; size_t off = 0;
  auto allocb = [&](size_t nbytes) -> char* {
    char* p = base + off;
    off += (nbytes + 255) & ~(size_t)255;
    return p;
  };
  auto alloc  = [&](size_t nfl) -> float*    { return (float*)allocb(nfl * 4); };
  auto alloch = [&](size_t nh)  -> _Float16* { return (_Float16*)allocb(nh * 2); };
  auto mk16 = [&](int pi, size_t n) -> const _Float16* {
    _Float16* d = alloch(n);
    cvt16<<<GRID(n), 256, 0, stream>>>(P(pi), d, (int)n);
    return d;
  };

  float *h[3], *c[3], *m[3], *kfs[3], *kfP[3], *flows[2];
  for (int l = 0; l < 3; ++l) { h[l] = alloc(NF); c[l] = alloc(NF); m[l] = alloc(NF); }
  for (int l = 0; l < 3; ++l) { kfs[l] = alloc(16 * 32); kfP[l] = alloc(16 * 1024); }
  flows[0] = alloc(NF); flows[1] = alloc(NF);
  float* xt    = alloc(NF);
  float* gates = alloc((size_t)4 * NF);
  float* wmout = alloc(NF);
  float* hlb   = alloc(NF);
  float* flbuf = alloc(NF);
  float* pooled= alloc(16 * 64 * 256);
  float* efeat = alloc(16 * 8192);
  float* elin  = alloc(16 * 32);
  float* eobs  = alloc(16 * 32);
  float* Fm    = alloc(1024);
  float* nlpre = alloc(16 * 128);
  float* nlln  = alloc(16 * 128);
  float* nl2   = alloc(16 * 32);
  float* spred = alloc(16 * 32);
  float* Ppred = alloc(16 * 1024);
  float* tmpA  = alloc(16 * 1024);
  float* Smat  = alloc(16 * 1024);
  float* Sinv  = alloc(16 * 1024);
  float* Kmat  = alloc(16 * 1024);
  float* innov = alloc(16 * 32);
  float* t1m   = alloc(16 * 1024);
  float* KRm   = alloc(16 * 1024);
  float* dvec  = alloc(16 * 8192);
  float* dres  = alloc(16 * 32 * 1024);
  float* doutb = alloc(NF);
  float* fbuf  = alloc(16 * 4096);
  float* xe1   = alloc(16 * 32 * 1024);
  float* dd1   = alloc(NF);
  float* ddT   = alloc(16 * 32 * 4096);
  float* decin = alloc(16 * 4096);

  // ---- one-time (per launch) f32 -> f16 weight conversion ----
  const _Float16* w_e1 = mk16(1, 288);
  const _Float16* w_e2 = mk16(3, 18432);
  const _Float16* w_fl[2] = { mk16(6, 8192), mk16(7, 8192) };
  const _Float16* w_d1 = mk16(91, 36864);
  const _Float16* w_d3 = mk16(95, 288);
  const _Float16 *wg16[3], *wm16[3], *encw16[3], *decw16[3];
  const _Float16 *enclw16[3], *declw16[3], *nlw1_16[3], *nlw2_16[3];
  for (int l = 0; l < 3; ++l) {
    int cb = CELL0 + CN * l;
    wg16[l]    = mk16(cb + 0, 442368);
    wm16[l]    = mk16(cb + 2, 36864);
    encw16[l]  = mk16(cb + 4, 18432);
    enclw16[l] = mk16(cb + 6, 262144);
    declw16[l] = mk16(cb + 10, 262144);
    decw16[l]  = mk16(cb + 12, 18432);
    nlw1_16[l] = mk16(cb + 18, 4096);
    nlw2_16[l] = mk16(cb + 22, 4096);
  }

  auto z = [&](float* p, int n) { fillz<<<GRID(n), 256, 0, stream>>>(p, n); };
  for (int l = 0; l < 3; ++l) {
    z(h[l], NF); z(c[l], NF); z(m[l], NF); z(kfs[l], 512);
    initP<<<GRID(16384), 256, 0, stream>>>(kfP[l], 16384);
  }
  z(flows[0], NF); z(flows[1], NF);

  auto convl = [&](const float* A, int Ca, const float* Bp, int Cb,
                   const float* Cp, int Cc, const _Float16* w, const float* bi,
                   float* o, int Hin, int Win, int Cout, int Hout, int Wout,
                   int ks, int st, int pd, int act, float sl) {
    ConvArgs ar;
    ar.inA = A; ar.inB = Bp; ar.inC = Cp; ar.Ca = Ca; ar.Cb = Cb; ar.Cc = Cc;
    ar.Cin = Ca + Cb + Cc; ar.w = w; ar.bias = bi; ar.out = o;
    ar.B = 16; ar.Hin = Hin; ar.Win = Win; ar.Cout = Cout;
    ar.hwshift = __builtin_ctz(Hout * Wout); ar.wshift = __builtin_ctz(Wout);
    ar.stride = st; ar.pad = pd; ar.act = act; ar.slope = sl;
    int M = 16 * Hout * Wout;
    dim3 g((M + 63) / 64, (Cout + 63) / 64);
    if (ks == 3) conv_wmma<3><<<g, 128, 0, stream>>>(ar);
    else         conv_wmma<1><<<g, 128, 0, stream>>>(ar);
  };
  auto gemm = [&](const float* X, const _Float16* W, const float* bi, float* Y,
                  int K, int N, int act, float sl) {
    rowgemm_wmma<<<dim3((N + 63) / 64), 128, 0, stream>>>(X, W, bi, Y, K, N, act, sl);
  };

  auto cell = [&](int l) {
    int cb = CELL0 + CN * l;
    // gates = conv(cat(xt,h,m)) ; c,m,h_lstm updates
    convl(xt, 64, h[l], 64, m[l], 64, wg16[l], P(cb + 1), gates,
          32, 32, 256, 32, 32, 3, 1, 1, ACT_NONE, 0.0f);
    lstm_cm<<<GRID(NF), 256, 0, stream>>>(gates, c[l], NF);
    convl(c[l], 64, nullptr, 0, nullptr, 0, wm16[l], P(cb + 3), wmout,
          32, 32, 64, 32, 32, 3, 1, 1, ACT_NONE, 0.0f);
    lstm_hm<<<GRID(NF), 256, 0, stream>>>(gates, wmout, c[l], m[l], hlb, NF);
    // measurement path
    pool2<<<GRID(16 * 64 * 256), 256, 0, stream>>>(hlb, pooled, 16 * 64 * 256);
    convl(pooled, 64, nullptr, 0, nullptr, 0, encw16[l], P(cb + 5), efeat,
          16, 16, 32, 16, 16, 3, 1, 1, ACT_LRELU, 0.01f);
    gemm(efeat, enclw16[l], P(cb + 7), elin, 8192, 32, ACT_NONE, 0.0f);
    layernorm_rows<<<16, 128, 0, stream>>>(elin, P(cb + 8), P(cb + 9), eobs, 32, 0, 0.0f);
    // Kalman step
    fm_build<<<1, 1024, 0, stream>>>(P(cb + 16), P(cb + 17), Fm);
    gemm(kfs[l], nlw1_16[l], P(cb + 19), nlpre, 32, 128, ACT_NONE, 0.0f);
    layernorm_rows<<<16, 128, 0, stream>>>(nlpre, P(cb + 20), P(cb + 21), nlln, 128, 1, 0.01f);
    gemm(nlln, nlw2_16[l], P(cb + 23), nl2, 128, 32, ACT_TANH, 0.0f);
    kf_pred<<<16, 32, 0, stream>>>(Fm, kfs[l], nl2, P(cb + 24), spred);
    bmm32<<<16, 1024, 0, stream>>>(Fm, 0, kfP[l], 1024, tmpA, 0, 0);     // Fm @ P
    bmm32<<<16, 1024, 0, stream>>>(tmpA, 1024, Fm, 0, Ppred, 1, 0);      // @ Fm^T
    add_diag<<<16, 32, 0, stream>>>(Ppred, P(cb + 25), 0.0f);            // + diag(Q)
    hipMemcpyAsync(Smat, Ppred, 16 * 1024 * sizeof(float), hipMemcpyDeviceToDevice, stream);
    add_diag<<<16, 32, 0, stream>>>(Smat, P(cb + 26), 1e-5f);            // + diag(R)+eps
    inv32<<<16, 1024, 0, stream>>>(Smat, Sinv);
    bmm32<<<16, 1024, 0, stream>>>(Ppred, 1024, Sinv, 1024, Kmat, 0, 0); // K gain
    ew_sub<<<GRID(512), 256, 0, stream>>>(eobs, spred, innov, 512);
    kf_supd<<<16, 32, 0, stream>>>(spred, Kmat, innov, kfs[l]);
    ew_t1<<<GRID(16384), 256, 0, stream>>>(Kmat, t1m, 16384);
    bmm32<<<16, 1024, 0, stream>>>(t1m, 1024, Ppred, 1024, tmpA, 0, 0);
    bmm32<<<16, 1024, 0, stream>>>(tmpA, 1024, t1m, 1024, kfP[l], 1, 0); // (I-K)P(I-K)^T
    ew_kr<<<GRID(16384), 256, 0, stream>>>(Kmat, P(cb + 26), KRm, 16384);
    bmm32<<<16, 1024, 0, stream>>>(KRm, 1024, Kmat, 1024, kfP[l], 1, 1); // + KRK^T
    // decoder path + h combine
    gemm(kfs[l], declw16[l], P(cb + 11), dvec, 32, 8192, ACT_LRELU, 0.01f);
    bilin<<<GRID(16 * 32 * 1024), 256, 0, stream>>>(dvec, dres, 16 * 32 * 1024);
    convl(dres, 32, nullptr, 0, nullptr, 0, decw16[l], P(cb + 13), doutb,
          32, 32, 64, 32, 32, 3, 1, 1, ACT_LRELU, 0.01f);
    combineh<<<GRID(NF), 256, 0, stream>>>(hlb, doutb, h[l], P(cb + 14), P(cb + 15), NF);
  };

  auto layer_loop = [&](int temb_t, bool do_flow_always, int t) {
    for (int l = 0; l < 3; ++l) {
      addb<<<GRID(NF), 256, 0, stream>>>(xt, temb + temb_t * 65536, NF);
      if (l > 0 && (do_flow_always || t > 0)) {
        convl(xt, 64, flows[l - 1], 64, nullptr, 0, w_fl[l - 1], P(8 + (l - 1)),
              flbuf, 32, 32, 64, 32, 32, 1, 1, 0, ACT_NONE, 0.0f);
        axpy<<<GRID(NF), 256, 0, stream>>>(xt, flbuf, 0.2f, NF);
        hipMemcpyAsync(flows[l - 1], flbuf, NF * sizeof(float), hipMemcpyDeviceToDevice, stream);
      }
      cell(l);
      hipMemcpyAsync(xt, h[l], NF * sizeof(float), hipMemcpyDeviceToDevice, stream);
    }
  };

  // -------- phase 1: warm up on 7 input frames --------
  for (int t = 0; t < 7; ++t) {
    gatherf<<<GRID(65536), 256, 0, stream>>>(x, fbuf, t);
    convl(fbuf, 1, nullptr, 0, nullptr, 0, w_e1, P(2), xe1,
          64, 64, 32, 32, 32, 3, 2, 1, ACT_LRELU, 0.2f);
    convl(xe1, 32, nullptr, 0, nullptr, 0, w_e2, P(4), xt,
          32, 32, 64, 32, 32, 3, 1, 1, ACT_LRELU, 0.2f);
    layer_loop(t, false, t);
  }
  // -------- phase 2: autoregressive 7-day forecast --------
  gatherf<<<GRID(65536), 256, 0, stream>>>(x, decin, 6);  // dec_in = x[:,-1]
  for (int t = 0; t < 7; ++t) {
    convl(decin, 1, nullptr, 0, nullptr, 0, w_e1, P(2), xe1,
          64, 64, 32, 32, 32, 3, 2, 1, ACT_LRELU, 0.2f);
    convl(xe1, 32, nullptr, 0, nullptr, 0, w_e2, P(4), xt,
          32, 32, 64, 32, 32, 3, 1, 1, ACT_LRELU, 0.2f);
    layer_loop(t + 7, true, t);
    // decode_frame
    convl(xt, 64, nullptr, 0, nullptr, 0, w_d1, P(92), dd1,
          32, 32, 64, 32, 32, 3, 1, 1, ACT_LRELU, 0.2f);
    convT_d2<<<GRID(16 * 32 * 4096), 256, 0, stream>>>(dd1, P(93), P(94), ddT, 0.2f);
    convl(ddT, 32, nullptr, 0, nullptr, 0, w_d3, P(96), decin,
          64, 64, 1, 64, 64, 3, 1, 1, ACT_SIG, 0.0f);
    scat<<<GRID(65536), 256, 0, stream>>>(decin, out, t);
  }
}
